// CausalSelfAttention_19774029431759
// MI455X (gfx1250) — matrix operand order repaired and checked by reading, not compiled, not run on hardware
//
#include <hip/hip_runtime.h>
#include <stdint.h>

// Problem constants (from the reference)
#define EMBED   1024
#define TOKENS  4096      // B*T
#define TSEQ    2048
#define HEADS   16
#define HDIM    64

typedef __attribute__((ext_vector_type(16))) _Float16 v16h;
typedef __attribute__((ext_vector_type(8)))  float    v8f;

union PackU { _Float16 h[2]; uint32_t u; };

__device__ __forceinline__ v16h ldfrag(const void* p0, const void* p1) {
  union { v16h h; uint4 q[2]; } u;
  u.q[0] = *(const uint4*)p0;
  u.q[1] = *(const uint4*)p1;
  return u.h;
}

__device__ __forceinline__ v8f wmma_f16(v16h a, v16h b, v8f c) {
  // (neg_a, A, neg_b, B, c_mod, C, reuse_a, reuse_b)
  return __builtin_amdgcn_wmma_f32_16x16x32_f16(false, a, false, b, (short)0, c,
                                                false, false);
}

__device__ __forceinline__ float rmax16(float x) {
#pragma unroll
  for (int m = 1; m < 16; m <<= 1) x = fmaxf(x, __shfl_xor(x, m, 32));
  return x;
}
__device__ __forceinline__ float rsum16(float x) {
#pragma unroll
  for (int m = 1; m < 16; m <<= 1) x += __shfl_xor(x, m, 32);
  return x;
}

// ---------------------------------------------------------------------------
// fp32 -> f16 elementwise convert
// ---------------------------------------------------------------------------
__global__ void k_f32_to_f16(const float* __restrict__ in,
                             _Float16* __restrict__ out, int n) {
  int i = blockIdx.x * 256 + threadIdx.x;
  if (i < n) out[i] = (_Float16)in[i];
}

// ---------------------------------------------------------------------------
// Convert+swizzle a 1024x1024 fp32 weight [K][N] into B-fragment order:
// dword index = ((kt*64 + nt)*32 + lane)*8 + v
// holds pack( W[kt*32 + 2v + 16*(lane>>4)][nt*16 + (lane&15)], W[k+1][n] )
// ---------------------------------------------------------------------------
__global__ void k_swizzle_w(const float* __restrict__ W,
                            uint32_t* __restrict__ out) {
  int id = blockIdx.x * 256 + threadIdx.x;       // 0 .. 524287
  int v  = id & 7;
  int L  = (id >> 3) & 31;
  int nt = (id >> 8) & 63;
  int kt = id >> 14;                              // 0..31
  int k  = kt * 32 + 2 * v + ((L >> 4) << 4);
  int n  = nt * 16 + (L & 15);
  PackU p;
  p.h[0] = (_Float16)W[(size_t)k * EMBED + n];
  p.h[1] = (_Float16)W[(size_t)(k + 1) * EMBED + n];
  out[id] = p.u;
}

// ---------------------------------------------------------------------------
// GEMM: C[4096x1024] = A(f16 row-major, 4096x1024) * Bsw(swizzled 1024x1024)
//       + bias, epilogue f16 or f32.
// block = 256 thr (8 waves), block tile 32x256, wave tile 16x64.
// ---------------------------------------------------------------------------
template <bool OUT_F16>
__global__ __launch_bounds__(256) void k_gemm(const _Float16* __restrict__ A,
                                              const uint32_t* __restrict__ Bsw,
                                              const float* __restrict__ bias,
                                              void* __restrict__ out) {
  const int tid  = threadIdx.x;
  const int wave = tid >> 5;
  const int lane = tid & 31;
  const int wm = wave & 1, wn = wave >> 1;
  const int m0 = blockIdx.x * 32 + wm * 16;
  const int n0 = blockIdx.y * 256 + wn * 64;
  const int row = lane & 15;
  const int hs  = lane >> 4;

  v8f acc[4];
#pragma unroll
  for (int ct = 0; ct < 4; ++ct)
#pragma unroll
    for (int v = 0; v < 8; ++v) acc[ct][v] = 0.0f;

  const _Float16* arow = A + (size_t)(m0 + row) * EMBED;

  for (int kt = 0; kt < 32; ++kt) {
    const int k0 = kt * 32;
    // Prefetch the A row and B fragment stream two k-steps ahead
    // (global_prefetch_b8; speculative, tail over-run is silently dropped).
    __builtin_prefetch(arow + k0 + 64, 0, 1);
    __builtin_prefetch(Bsw + (((size_t)(kt + 2) * 64 + (n0 >> 4)) * 32 + lane) * 8,
                       0, 1);
    v16h af = ldfrag(arow + k0 + hs * 16, arow + k0 + hs * 16 + 8);
#pragma unroll
    for (int ct = 0; ct < 4; ++ct) {
      const int nt = (n0 >> 4) + ct;
      const uint32_t* bp = Bsw + (((size_t)kt * 64 + nt) * 32 + lane) * 8;
      v16h bf = ldfrag(bp, bp + 4);
      acc[ct] = wmma_f16(af, bf, acc[ct]);
    }
  }

#pragma unroll
  for (int ct = 0; ct < 4; ++ct) {
    const int n = n0 + ct * 16 + (lane & 15);
    const float bv = bias[n];
#pragma unroll
    for (int v = 0; v < 8; ++v) {
      const int m = m0 + v + 8 * hs;
      const float val = acc[ct][v] + bv;
      if (OUT_F16)
        ((_Float16*)out)[(size_t)m * EMBED + n] = (_Float16)val;
      else
        ((float*)out)[(size_t)m * EMBED + n] = val;
    }
  }
}

// ---------------------------------------------------------------------------
// Flash attention (causal). grid = (T/128, H, B), block = 256 (8 waves).
// Each wave owns 16 query rows; streams 64-key tiles with online softmax.
// ---------------------------------------------------------------------------
__global__ __launch_bounds__(256) void k_flash(const _Float16* __restrict__ Qh,
                                               const _Float16* __restrict__ Kh,
                                               const _Float16* __restrict__ Vh,
                                               _Float16* __restrict__ Oh) {
  const int tid  = threadIdx.x;
  const int wave = tid >> 5;
  const int lane = tid & 31;
  const int row  = lane & 15;
  const int hs   = lane >> 4;

  const int    m0      = blockIdx.x * 128 + wave * 16;  // first q row of wave
  const size_t tokbase = (size_t)blockIdx.z * TSEQ;
  const int    hcol    = blockIdx.y * HDIM;

  // V tile staged in B-fragment order: [ks(2)][nt(4)][lane(32)][v(8)] dwords
  __shared__ __align__(16) uint32_t vbuf[2 * 4 * 32 * 8];   // 8 KB
  // P tiles, one 16x64 f16 tile per wave, row stride 72 f16 (144 B = 9*16 B)
  __shared__ __align__(16) _Float16 pbuf[8][16 * 72];       // 18 KB

  // Q fragments resident in registers (hd = 64 -> two K=32 fragments)
  const _Float16* qrow = Qh + (tokbase + m0 + row) * EMBED + hcol;
  v16h qf0 = ldfrag(qrow + hs * 16,      qrow + hs * 16 + 8);
  v16h qf1 = ldfrag(qrow + 32 + hs * 16, qrow + 32 + hs * 16 + 8);

  float mrow[8], lrow[8];
  v8f o[4];
#pragma unroll
  for (int v = 0; v < 8; ++v) { mrow[v] = -1.0e30f; lrow[v] = 0.0f; }
#pragma unroll
  for (int nt = 0; nt < 4; ++nt)
#pragma unroll
    for (int v = 0; v < 8; ++v) o[nt][v] = 0.0f;

  const int ntiles = blockIdx.x * 2 + 2;   // key tiles needed by this block

  for (int kt = 0; kt < ntiles; ++kt) {
    // ---- cooperatively stage V tile (64 keys x 64 dims) into frag order ----
    __syncthreads();
    for (int i = tid; i < 2048; i += 256) {
      const int v  = i & 7;
      const int L  = (i >> 3) & 31;
      const int nt = (i >> 8) & 3;
      const int ks = (i >> 10) & 1;
      const int kk = kt * 64 + ks * 32 + 2 * v + 16 * (L >> 4);
      const int d  = nt * 16 + (L & 15);
      const _Float16* vp = Vh + (tokbase + kk) * EMBED + hcol + d;
      // Prefetch the same addresses for the next key tile.
      if (kt + 1 < ntiles) __builtin_prefetch(vp + (size_t)64 * EMBED, 0, 1);
      PackU p;
      p.h[0] = vp[0];
      p.h[1] = vp[EMBED];
      vbuf[i] = p.u;
    }
    __syncthreads();

    if (kt * 64 <= m0 + 15) {      // wave has at least one unmasked element
      // ---- S = (Q K^T) * scale, causal mask ----
      v8f s[4];
#pragma unroll
      for (int ct = 0; ct < 4; ++ct) {
#pragma unroll
        for (int v = 0; v < 8; ++v) s[ct][v] = 0.0f;
        const int kkcol = kt * 64 + ct * 16 + (lane & 15);
        const _Float16* krow = Kh + (tokbase + kkcol) * EMBED + hcol;
        // Prefetch next key tile's K fragment row.
        __builtin_prefetch(krow + (size_t)64 * EMBED, 0, 1);
        v16h bf0 = ldfrag(krow + hs * 16,      krow + hs * 16 + 8);
        v16h bf1 = ldfrag(krow + 32 + hs * 16, krow + 32 + hs * 16 + 8);
        s[ct] = wmma_f16(qf0, bf0, s[ct]);
        s[ct] = wmma_f16(qf1, bf1, s[ct]);
      }

      const bool partial = (kt * 64 + 63) > m0;
#pragma unroll
      for (int ct = 0; ct < 4; ++ct) {
        const int kcol = kt * 64 + ct * 16 + (lane & 15);
#pragma unroll
        for (int v = 0; v < 8; ++v) {
          float xv = s[ct][v] * 0.125f;            // 1/sqrt(64)
          if (partial && (kcol > m0 + v + 8 * hs)) xv = -1.0e30f;
          s[ct][v] = xv;
        }
      }

      // ---- online softmax ----
      float corr[8];
#pragma unroll
      for (int v = 0; v < 8; ++v) {
        float t = fmaxf(fmaxf(s[0][v], s[1][v]), fmaxf(s[2][v], s[3][v]));
        t = rmax16(t);
        const float mn = fmaxf(mrow[v], t);
        corr[v] = __expf(mrow[v] - mn);
        mrow[v] = mn;
      }

      _Float16* pb = &pbuf[wave][0];
      float rs[8];
#pragma unroll
      for (int v = 0; v < 8; ++v) rs[v] = 0.0f;
#pragma unroll
      for (int ct = 0; ct < 4; ++ct) {
#pragma unroll
        for (int v = 0; v < 8; ++v) {
          const float p = __expf(s[ct][v] - mrow[v]);
          rs[v] += p;
          pb[(v + 8 * hs) * 72 + ct * 16 + (lane & 15)] = (_Float16)p;
        }
      }
#pragma unroll
      for (int v = 0; v < 8; ++v) {
        lrow[v] = lrow[v] * corr[v] + rsum16(rs[v]);
#pragma unroll
        for (int nt = 0; nt < 4; ++nt) o[nt][v] *= corr[v];
      }

      __threadfence_block();   // compiler fence; same-wave DS ops are in-order

      // ---- O += P * V ----
#pragma unroll
      for (int ks = 0; ks < 2; ++ks) {
        const _Float16* pr = pb + (lane & 15) * 72 + ks * 32;
        v16h pa = ldfrag(pr + hs * 16, pr + hs * 16 + 8);
#pragma unroll
        for (int nt = 0; nt < 4; ++nt) {
          const uint32_t* vp = &vbuf[((ks * 4 + nt) * 32 + lane) * 8];
          v16h vf = ldfrag(vp, vp + 4);
          o[nt] = wmma_f16(pa, vf, o[nt]);
        }
      }
    }
  }

  // ---- normalize and store (f16, row-major [token][embed]) ----
#pragma unroll
  for (int v = 0; v < 8; ++v) {
    const float inv = 1.0f / lrow[v];
    const size_t trow = (tokbase + m0 + v + 8 * hs) * EMBED + hcol;
#pragma unroll
    for (int nt = 0; nt < 4; ++nt)
      Oh[trow + nt * 16 + (lane & 15)] = (_Float16)(o[nt][v] * inv);
  }
}

// ---------------------------------------------------------------------------
// Host side
// ---------------------------------------------------------------------------
extern "C" void kernel_launch(void* const* d_in, const int* in_sizes, int n_in,
                              void* d_out, int out_size, void* d_ws,
                              size_t ws_size, hipStream_t stream) {
  (void)in_sizes; (void)n_in; (void)out_size; (void)ws_size;

  const float* x  = (const float*)d_in[0];
  const float* Wq = (const float*)d_in[1];
  const float* bq = (const float*)d_in[2];
  const float* Wk = (const float*)d_in[3];
  const float* bk = (const float*)d_in[4];
  const float* Wv = (const float*)d_in[5];
  const float* bv = (const float*)d_in[6];
  const float* Wo = (const float*)d_in[7];
  const float* bo = (const float*)d_in[8];

  // Workspace layout (48 MB total)
  char* ws = (char*)d_ws;
  const size_t MB = 1024ull * 1024ull;
  _Float16* xh   = (_Float16*)(ws + 0);        // 8 MB
  uint32_t* Wqsw = (uint32_t*)(ws + 8 * MB);   // 2 MB each
  uint32_t* Wksw = (uint32_t*)(ws + 10 * MB);
  uint32_t* Wvsw = (uint32_t*)(ws + 12 * MB);
  uint32_t* Wosw = (uint32_t*)(ws + 14 * MB);
  _Float16* Qh   = (_Float16*)(ws + 16 * MB);  // 8 MB each
  _Float16* Kh   = (_Float16*)(ws + 24 * MB);
  _Float16* Vh   = (_Float16*)(ws + 32 * MB);
  _Float16* Ah   = (_Float16*)(ws + 40 * MB);

  const int NX = TOKENS * EMBED;               // 4,194,304
  k_f32_to_f16<<<NX / 256, 256, 0, stream>>>(x, xh, NX);

  const int NSW = EMBED * EMBED / 2 / 256;     // 2048 blocks
  k_swizzle_w<<<NSW, 256, 0, stream>>>(Wq, Wqsw);
  k_swizzle_w<<<NSW, 256, 0, stream>>>(Wk, Wksw);
  k_swizzle_w<<<NSW, 256, 0, stream>>>(Wv, Wvsw);
  k_swizzle_w<<<NSW, 256, 0, stream>>>(Wo, Wosw);

  dim3 ggrid(TOKENS / 32, EMBED / 256);        // 128 x 4
  k_gemm<true><<<ggrid, 256, 0, stream>>>(xh, Wqsw, bq, (void*)Qh);
  k_gemm<true><<<ggrid, 256, 0, stream>>>(xh, Wksw, bk, (void*)Kh);
  k_gemm<true><<<ggrid, 256, 0, stream>>>(xh, Wvsw, bv, (void*)Vh);

  dim3 fgrid(TSEQ / 128, HEADS, 2);            // 16 x 16 x 2
  k_flash<<<fgrid, 256, 0, stream>>>(Qh, Kh, Vh, Ah);

  k_gemm<false><<<ggrid, 256, 0, stream>>>(Ah, Wosw, bo, d_out);
}
